// DeformBasicBlock_6313601925210
// MI455X (gfx1250) — matrix operand order repaired and checked by reading, not compiled, not run on hardware
//
#include <hip/hip_runtime.h>
#include <cstddef>
#include <cstdint>

#define Pp    25088      // D*H*W
#define NQ    50176      // N*P
#define KK    1728       // 27 taps * 64 ch
#define KSTEPS 54        // 1728/32
#define NPOS  32         // positions per block
#define LDSK  1736       // padded row stride (halves) for im2col tile
#define VTS   72         // padded row stride (halves) for per-tap sample tile
#define OFFC  648
#define OFFCP 704        // 44 tiles of 16 -> 11 tiles per wave, uniform

#define SMEM_BYTES (NPOS * LDSK * 2)          // 111,104 B dynamic LDS
#define OFFLDS_B   (OFFC * NPOS * 4)          //  82,944 B (overlays im2col)

typedef _Float16 half_t;
typedef __attribute__((ext_vector_type(16))) _Float16 v16h;
typedef __attribute__((ext_vector_type(8)))  float    v8f;
typedef __attribute__((ext_vector_type(4)))  float    vf4;

union HVec { vf4 f4[2]; v16h v; };

extern __shared__ unsigned char smem_dyn[];

// A-matrix (16x32 f16): lane m=l&15, kbase8=(l>>4)*8,
// halves j=0..7 -> K=kbase8+j ; j=8..15 -> K=16+kbase8+(j-8)
__device__ __forceinline__ v16h load_a_tile(const half_t* __restrict__ wb,
                                            int obase, int kb, int lane) {
  const int m   = lane & 15;
  const int kb8 = (lane >> 4) << 3;
  const half_t* p = wb + (size_t)(obase + m) * KK + kb + kb8;
  HVec u;
  u.f4[0] = *(const vf4*)(p);
  u.f4[1] = *(const vf4*)(p + 16);
  return u.v;
}

// B-matrix (32x16 f16): lane col n=l&15, kbase8=(l>>4)*8,
// halves j=0..7 -> K=kbase8+j ; j=8..15 -> K=16+kbase8+(j-8)
__device__ __forceinline__ v16h load_b_tile(const half_t* base, int rstride,
                                            int kb, int lane) {
  const int n  = lane & 15;
  const int k0 = kb + ((lane >> 4) << 3);
  const half_t* p = base + n * rstride + k0;
  HVec u;
  u.f4[0] = *(const vf4*)(p);
  u.f4[1] = *(const vf4*)(p + 16);
  return u.v;
}

// im2col of NPOS positions into LDS via CDNA5 async global->LDS DMA (ASYNCcnt).
// Valid taps: 8x global_load_async_to_lds_b128 (16B/lane, memory->LDS direct).
// Pad taps: plain LDS zero stores. Caller barriers after; we fence ASYNCcnt here.
__device__ __forceinline__ void build_im2col(half_t* B0, const half_t* __restrict__ xb,
                                             int q0, int tid) {
  for (int u = tid; u < NPOS * 27; u += 256) {
    const int i = u / 27, t = u - i * 27;
    const int q = q0 + i;
    const int n = q / Pp;       const int p  = q - n * Pp;
    const int d = p / 3136;     const int rm = p - d * 3136;
    const int h = rm / 56;      const int w  = rm - h * 56;
    const int kd = t / 9; const int r2 = t - kd * 9;
    const int kh = r2 / 3; const int kw = r2 - kh * 3;
    const int sd = d + kd - 1, sh = h + kh - 1, sw = w + kw - 1;
    half_t* dst = B0 + i * LDSK + t * 64;
    if ((unsigned)sd < 8u && (unsigned)sh < 56u && (unsigned)sw < 56u) {
      const int sp = (sd * 56 + sh) * 56 + sw;
      const half_t* src = xb + ((size_t)n * Pp + sp) * 64;
#pragma unroll
      for (int j = 0; j < 8; ++j) {
        const unsigned ldsoff = (unsigned)(uintptr_t)(dst + j * 8);
        const unsigned long long ga = (unsigned long long)(uintptr_t)(src + j * 8);
        asm volatile("global_load_async_to_lds_b128 %0, %1, off"
                     :: "v"(ldsoff), "v"(ga) : "memory");
      }
    } else {
      vf4 z = {0.f, 0.f, 0.f, 0.f};
#pragma unroll
      for (int j = 0; j < 8; ++j) ((vf4*)dst)[j] = z;
    }
  }
  asm volatile("s_wait_asynccnt 0" ::: "memory");
}

// ---------------- pack kernels ----------------
__global__ void pack_x_kernel(const float* __restrict__ x, half_t* __restrict__ xb) {
  const int idx = blockIdx.x * blockDim.x + threadIdx.x;
  if (idx >= 64 * NQ) return;
  const int n = idx / (64 * Pp); const int rm = idx - n * (64 * Pp);
  const int c = rm / Pp;         const int p  = rm - c * Pp;
  xb[((size_t)n * Pp + p) * 64 + c] = (half_t)x[idx];
}

// w[O][64][27] f32 -> wb[Opad][27*64] f16 with K = t*64 + c (zero-padded rows)
__global__ void pack_w_kernel(const float* __restrict__ src, half_t* __restrict__ dst,
                              int O, int total) {
  const int idx = blockIdx.x * blockDim.x + threadIdx.x;
  if (idx >= total) return;
  const int o = idx / KK; const int k = idx - o * KK;
  const int t = k >> 6;   const int c = k & 63;
  const float v = (o < O) ? src[((size_t)o * 64 + c) * 27 + t] : 0.f;
  dst[idx] = (half_t)v;
}

// ---- fused: offset-conv (WMMA) + deformable sampling + deform GEMM (WMMA) ----
// 256 threads = 8 waves; 32 positions/block.
// GEMM1: wave w -> N-tile (w&1), O-strip (w>>1); 11 O-tiles each (44 total, padded).
// GEMM2: wave w -> O-tile (w&3), N-tile (w>>2); one WMMA tile per wave, no divergence.
__global__ __launch_bounds__(256, 1) void deform_fused_kernel(
    const half_t* __restrict__ xb, const half_t* __restrict__ w_offb,
    const float* __restrict__ b_off, const half_t* __restrict__ w1b,
    float* __restrict__ out1) {
  half_t* B0      = (half_t*)smem_dyn;                 // NPOS*LDSK*2 (phases 1-2)
  float*  off_lds = (float*)smem_dyn;                  // 648*32*4 (phase 3, overlays B0)
  half_t* Vt      = (half_t*)(smem_dyn + OFFLDS_B);    // 32*VTS*2 = 4608 B

  const int tid = threadIdx.x;
  const int lane = tid & 31;
  const int wave = tid >> 5;
  const int q0 = blockIdx.x * NPOS;

  build_im2col(B0, xb, q0, tid);
  __syncthreads();

  // ---- GEMM1: 704x32 offsets over K=1728, fully uniform tiling ----
  const int nt1 = wave & 1;        // N-tile (16 positions)
  const int st1 = wave >> 1;       // O-strip 0..3
  const half_t* B0n = B0 + nt1 * 16 * LDSK;
  v8f acc[11] = {};
  for (int s = 0; s < KSTEPS; ++s) {
    const int kb = s << 5;
    v16h b = load_b_tile(B0n, LDSK, kb, lane);
#pragma unroll
    for (int r = 0; r < 11; ++r) {
      const int ti = r * 4 + st1;  // 0..43, all valid (padded weights)
      v16h a = load_a_tile(w_offb, ti << 4, kb, lane);
      acc[r] = __builtin_amdgcn_wmma_f32_16x16x32_f16(false, a, false, b,
                                                      (short)0, acc[r], false, false);
    }
  }
  __syncthreads();   // B0 dead; reuse as off_lds

  {
    const int hi = (lane >> 4) << 3;
    const int nidx = lane & 15;
#pragma unroll
    for (int r = 0; r < 11; ++r) {
      const int ti = r * 4 + st1;
#pragma unroll
      for (int rr = 0; rr < 8; ++rr) {
        const int oc = (ti << 4) + rr + hi;
        if (oc < OFFC)
          off_lds[oc * NPOS + nt1 * 16 + nidx] = acc[r][rr] + b_off[oc];
      }
    }
  }
  __syncthreads();

  // ---- Phase 3: per-tap trilinear sampling + WMMA consume ----
  const int si = tid >> 3;         // position 0..31
  const int sg = tid & 7;          // offset group 0..7 (8 channels each)
  const int q = q0 + si;
  const int n = q / Pp;   const int p  = q - n * Pp;
  const int d = p / 3136; const int rm = p - d * 3136;
  const int h = rm / 56;  const int w  = rm - h * 56;

  const int ot2 = wave & 3;        // GEMM2 O-tile
  const int nt2 = wave >> 2;       // GEMM2 N-tile
  const half_t* VtN = Vt + nt2 * 16 * VTS;

  v8f acc2 = {};
  for (int t = 0; t < 27; ++t) {
    const int kd = t / 9; const int r2 = t - kd * 9;
    const int kh = r2 / 3; const int kw = r2 - kh * 3;
    const int ob = ((sg * 27 + t) * 3) * NPOS;
    const float od = off_lds[ob + si];
    const float oh = off_lds[ob + NPOS + si];
    const float ow = off_lds[ob + 2 * NPOS + si];
    const float pd = (float)(d + kd - 1) + od;
    const float ph = (float)(h + kh - 1) + oh;
    const float pw = (float)(w + kw - 1) + ow;
    const float fd0 = floorf(pd), fh0 = floorf(ph), fw0 = floorf(pw);
    const int d0 = (int)fd0, h0 = (int)fh0, w0 = (int)fw0;
    const float fd = pd - fd0, fh = ph - fh0, fw = pw - fw0;
    float a[8] = {0.f, 0.f, 0.f, 0.f, 0.f, 0.f, 0.f, 0.f};
#pragma unroll
    for (int cd = 0; cd < 2; ++cd)
#pragma unroll
      for (int ch = 0; ch < 2; ++ch)
#pragma unroll
        for (int cw = 0; cw < 2; ++cw) {
          const int id = d0 + cd, ih = h0 + ch, iw = w0 + cw;
          if ((unsigned)id < 8u && (unsigned)ih < 56u && (unsigned)iw < 56u) {
            const float wgt = (cd ? fd : 1.f - fd) * (ch ? fh : 1.f - fh)
                            * (cw ? fw : 1.f - fw);
            const int sp = (id * 56 + ih) * 56 + iw;
            union { vf4 f4; half_t hh[8]; } uu;
            uu.f4 = *(const vf4*)(xb + ((size_t)n * Pp + sp) * 64 + sg * 8);
#pragma unroll
            for (int j = 0; j < 8; ++j) a[j] += wgt * (float)uu.hh[j];
          }
        }
    __syncthreads();    // previous tap's WMMA reads of Vt complete
    {
      union { vf4 f4; half_t hh[8]; } pk;
#pragma unroll
      for (int j = 0; j < 8; ++j) pk.hh[j] = (half_t)a[j];
      *(vf4*)(Vt + si * VTS + sg * 8) = pk.f4;
    }
    __syncthreads();
    // one (O-tile, N-tile) per wave: no divergence around WMMA
#pragma unroll
    for (int s = 0; s < 2; ++s) {
      v16h b = load_b_tile(VtN, VTS, s << 5, lane);
      v16h av = load_a_tile(w1b, ot2 << 4, (t << 6) + (s << 5), lane);
      acc2 = __builtin_amdgcn_wmma_f32_16x16x32_f16(false, av, false, b,
                                                    (short)0, acc2, false, false);
    }
  }

  {
    const int hi = (lane >> 4) << 3;
    const int nidx = lane & 15;
    const int qq = q0 + nt2 * 16 + nidx;
    const int nn = qq / Pp; const int pp = qq - nn * Pp;
#pragma unroll
    for (int rr = 0; rr < 8; ++rr) {
      const int o = (ot2 << 4) + rr + hi;
      out1[((size_t)nn * 64 + o) * Pp + pp] = acc2[rr];
    }
  }
}

// ---------------- plain conv (64 out ch) via WMMA, 32 pos/block, 8 uniform wave-tiles ----
__global__ __launch_bounds__(256, 1) void conv_gemm64_kernel(
    const half_t* __restrict__ xin, const half_t* __restrict__ wb,
    const float* __restrict__ bias, float* __restrict__ outp) {
  half_t* B0 = (half_t*)smem_dyn;
  const int tid = threadIdx.x, lane = tid & 31, wave = tid >> 5;
  const int q0 = blockIdx.x * NPOS;
  build_im2col(B0, xin, q0, tid);
  __syncthreads();
  const int ot = wave & 3;
  const int nt = wave >> 2;
  const half_t* B0n = B0 + nt * 16 * LDSK;
  v8f acc = {};
  for (int s = 0; s < KSTEPS; ++s) {
    const int kb = s << 5;
    v16h b = load_b_tile(B0n, LDSK, kb, lane);
    v16h a = load_a_tile(wb, ot << 4, kb, lane);
    acc = __builtin_amdgcn_wmma_f32_16x16x32_f16(false, a, false, b,
                                                 (short)0, acc, false, false);
  }
  const int hi = (lane >> 4) << 3, nidx = lane & 15;
  const int qq = q0 + nt * 16 + nidx;
  const int nn = qq / Pp; const int pp = qq - nn * Pp;
#pragma unroll
  for (int rr = 0; rr < 8; ++rr) {
    const int o = (ot << 4) + rr + hi;
    outp[((size_t)nn * 64 + o) * Pp + pp] = acc[rr] + bias[o];
  }
}

// ---------------- batchnorm: deterministic two-stage reduction ----------------
// Stage 1: grid = 64 channels * 8 slices; per-block tree reduce -> partial sums.
__global__ __launch_bounds__(256) void bn_partial_kernel(const float* __restrict__ src,
                                                         float* __restrict__ part) {
  const int c = blockIdx.x >> 3, sl = blockIdx.x & 7, tid = threadIdx.x;
  const int i0 = sl * (Pp / 8);            // 3136-wide slice
  float s = 0.f, s2 = 0.f;
  for (int n = 0; n < 2; ++n) {
    const float* p = src + ((size_t)n * 64 + c) * Pp + i0;
    for (int i = tid; i < Pp / 8; i += 256) { const float v = p[i]; s += v; s2 += v * v; }
  }
  __shared__ float sh[256], sh2[256];
  sh[tid] = s; sh2[tid] = s2; __syncthreads();
  for (int o = 128; o > 0; o >>= 1) {
    if (tid < o) { sh[tid] += sh[tid + o]; sh2[tid] += sh2[tid + o]; }
    __syncthreads();
  }
  if (tid == 0) {
    part[(c * 8 + sl) * 2 + 0] = sh[0];
    part[(c * 8 + sl) * 2 + 1] = sh2[0];
  }
}

// Stage 2: one tiny block; fixed-order sum over 8 partials per channel.
__global__ void bn_finalize_kernel(const float* __restrict__ part,
                                   float* __restrict__ mean, float* __restrict__ rstd) {
  const int c = threadIdx.x;
  if (c >= 64) return;
  float s = 0.f, s2 = 0.f;
  for (int sl = 0; sl < 8; ++sl) {
    s  += part[(c * 8 + sl) * 2 + 0];
    s2 += part[(c * 8 + sl) * 2 + 1];
  }
  const float m = s / (float)NQ;
  const float v = s2 / (float)NQ - m * m;
  mean[c] = m; rstd[c] = rsqrtf(v + 1e-5f);
}

__global__ void bn_relu_pack_kernel(const float* __restrict__ src,
    const float* __restrict__ mean, const float* __restrict__ rstd,
    const float* __restrict__ gamma, const float* __restrict__ beta,
    half_t* __restrict__ dst) {
  const int idx = blockIdx.x * blockDim.x + threadIdx.x;
  if (idx >= 64 * NQ) return;
  const int n = idx / (64 * Pp); const int rm = idx - n * (64 * Pp);
  const int c = rm / Pp;         const int p  = rm - c * Pp;
  float v = gamma[c] * (src[idx] - mean[c]) * rstd[c] + beta[c];
  v = fmaxf(v, 0.f);
  dst[((size_t)n * Pp + p) * 64 + c] = (half_t)v;
}

__global__ void bn_add_relu_kernel(const float* __restrict__ src,
    const float* __restrict__ mean, const float* __restrict__ rstd,
    const float* __restrict__ gamma, const float* __restrict__ beta,
    const float* __restrict__ resid, float* __restrict__ outp) {
  const int idx = blockIdx.x * blockDim.x + threadIdx.x;
  if (idx >= 64 * NQ) return;
  const int c = (idx / Pp) & 63;
  const float v = gamma[c] * (src[idx] - mean[c]) * rstd[c] + beta[c] + resid[idx];
  outp[idx] = fmaxf(v, 0.f);
}

// ---------------- host launch ----------------
extern "C" void kernel_launch(void* const* d_in, const int* in_sizes, int n_in,
                              void* d_out, int out_size, void* d_ws, size_t ws_size,
                              hipStream_t stream) {
  (void)in_sizes; (void)n_in; (void)out_size; (void)ws_size;
  const float* x      = (const float*)d_in[0];
  const float* w_off  = (const float*)d_in[1];
  const float* b_off  = (const float*)d_in[2];
  const float* w1     = (const float*)d_in[3];
  const float* gamma1 = (const float*)d_in[4];
  const float* beta1  = (const float*)d_in[5];
  const float* w2     = (const float*)d_in[6];
  const float* b2     = (const float*)d_in[7];
  const float* gamma2 = (const float*)d_in[8];
  const float* beta2  = (const float*)d_in[9];
  float* outp = (float*)d_out;

  char* ws = (char*)d_ws;
  half_t* xb     = (half_t*)(ws + 0);           //  6,422,528 B
  half_t* xb2    = (half_t*)(ws + 6422528);     //  6,422,528 B
  half_t* w_offb = (half_t*)(ws + 12845056);    //  2,433,024 B (704 padded rows)
  half_t* w1b    = (half_t*)(ws + 15278080);    //    221,184 B
  half_t* w2b    = (half_t*)(ws + 15499264);    //    221,184 B
  float*  out1   = (float*)(ws + 15720448);     // 12,845,056 B
  float*  out2   = (float*)(ws + 28565504);     // 12,845,056 B
  float*  stats  = (float*)(ws + 41410560);     // mean1,rstd1,mean2,rstd2 (4*64 f32)
  float*  part   = (float*)(ws + 41411584);     // 64*8*2 f32 partials (4,096 B)

  const int NE = 64 * NQ;                 // 3,211,264
  pack_x_kernel<<<(NE + 255) / 256, 256, 0, stream>>>(x, xb);
  pack_w_kernel<<<(OFFCP * KK + 255) / 256, 256, 0, stream>>>(w_off, w_offb, OFFC, OFFCP * KK);
  pack_w_kernel<<<(64 * KK + 255) / 256, 256, 0, stream>>>(w1, w1b, 64, 64 * KK);
  pack_w_kernel<<<(64 * KK + 255) / 256, 256, 0, stream>>>(w2, w2b, 64, 64 * KK);

  deform_fused_kernel<<<NQ / NPOS, 256, SMEM_BYTES, stream>>>(xb, w_offb, b_off, w1b, out1);

  bn_partial_kernel<<<64 * 8, 256, 0, stream>>>(out1, part);
  bn_finalize_kernel<<<1, 64, 0, stream>>>(part, stats + 0, stats + 64);
  bn_relu_pack_kernel<<<(NE + 255) / 256, 256, 0, stream>>>(out1, stats + 0, stats + 64,
                                                            gamma1, beta1, xb2);

  conv_gemm64_kernel<<<NQ / NPOS, 256, SMEM_BYTES, stream>>>(xb2, w2b, b2, out2);

  bn_partial_kernel<<<64 * 8, 256, 0, stream>>>(out2, part);
  bn_finalize_kernel<<<1, 64, 0, stream>>>(part, stats + 128, stats + 192);
  bn_add_relu_kernel<<<(NE + 255) / 256, 256, 0, stream>>>(out2, stats + 128, stats + 192,
                                                           gamma2, beta2, x, outp);
}
